// SequentialFeedForward_78486232367320
// MI455X (gfx1250) — matrix operand order repaired and checked by reading, not compiled, not run on hardware
//
#include <hip/hip_runtime.h>
#include <hip/hip_bf16.h>

#define SEQ_L   2048
#define WIDTH   2048
#define WS      32
#define NUM_AA  21
#define IN_SZ   704          // WS*NUM_AA + WS
#define NWG     64           // persistent grid: must be co-resident
#define TPB     256          // 8 waves (wave32)
#define LAYER_ELEMS (WIDTH*WIDTH)   // 4M = 2^22

typedef __attribute__((ext_vector_type(16))) _Float16 v16h;
typedef __attribute__((ext_vector_type(8)))  _Float16 v8h;
typedef __attribute__((ext_vector_type(8)))  float    v8f;

// ---------------------------------------------------------------- barrier init
__global__ void init_bar_kernel(unsigned* cnt, unsigned* gen) {
    *cnt = 0u; *gen = 0u;
}

// -------------------------------------------------- precompute A0[t][r] (one-hot gather)
// A0[t][r] = sum_{j=0..31, t-31+j>=0} W0[r*704 + j*21 + x[t-31+j]]
__global__ void precompute_A0_kernel(const int* __restrict__ x,
                                     const float* __restrict__ W0,
                                     float* __restrict__ A0) {
    __shared__ int xs[WS];
    int t = blockIdx.x;
    if (threadIdx.x < WS) {
        int si = t - (WS - 1) + threadIdx.x;
        xs[threadIdx.x] = (si >= 0) ? x[si] : -1;
    }
    __syncthreads();
    for (int rr = 0; rr < WIDTH / TPB; ++rr) {
        int r = rr * TPB + threadIdx.x;
        const float* row = W0 + (size_t)r * IN_SZ;
        float s = 0.f;
        #pragma unroll
        for (int j = 0; j < WS; ++j) {
            int c = xs[j];
            if (c >= 0) s += row[j * NUM_AA + c];
        }
        A0[(size_t)t * WIDTH + r] = s;
    }
}

// ------------------------- convert W1..W4 f32 -> f16, WMMA *B-operand* tile swizzle
// B operand of V_WMMA_F32_16X16X32_F16 is a 32x16 f16 matrix:
//   lane%16 = output column N (0..15), lane/16 = h, element s -> K = 8h+s (s<8), 16+8h+(s-8) (s>=8)
// We want D[.][n] = sum_k h[k0+k] * W[rt*16+n][k0+k], so:
//   element(block=(rt,kc), lane=16*((kk>>3)&1)+n, s=(kk&7)+((kk>>4)<<3)) = W[rt*16 + n][kc*32 + kk]
__global__ void convert_swizzle_kernel(const float* __restrict__ W1,
                                       const float* __restrict__ W2,
                                       const float* __restrict__ W3,
                                       const float* __restrict__ W4,
                                       _Float16* __restrict__ Wsw) {
    size_t e = (size_t)blockIdx.x * blockDim.x + threadIdx.x;   // < 4*2^22
    int layer = (int)(e >> 22);
    size_t rem = e & (size_t)(LAYER_ELEMS - 1);
    int R = (int)(rem >> 11);
    int K = (int)(rem & (WIDTH - 1));
    const float* Wl = (layer == 0) ? W1 : (layer == 1) ? W2 : (layer == 2) ? W3 : W4;
    float v = Wl[(size_t)R * WIDTH + K];
    int rt = R >> 4, n = R & 15, kc = K >> 5, kk = K & 31;
    size_t dst = ((size_t)layer << 22) + ((size_t)(rt * 64 + kc)) * 512
               + (size_t)(((kk >> 3) & 1) * 16 + n) * 16 + (kk & 7) + ((kk >> 4) << 3);
    Wsw[dst] = (_Float16)v;
}

// ---------------------------------------------------------------- device barrier
__device__ __forceinline__ void gbarrier(unsigned* cnt, unsigned* gen) {
    __syncthreads();
    if (threadIdx.x == 0) {
        __threadfence();   // make this WG's global stores visible device-wide
        unsigned g = __hip_atomic_load(gen, __ATOMIC_RELAXED, __HIP_MEMORY_SCOPE_AGENT);
        unsigned a = __hip_atomic_fetch_add(cnt, 1u, __ATOMIC_ACQ_REL, __HIP_MEMORY_SCOPE_AGENT);
        if (a == NWG - 1) {
            __hip_atomic_store(cnt, 0u, __ATOMIC_RELAXED, __HIP_MEMORY_SCOPE_AGENT);
            __hip_atomic_store(gen, g + 1u, __ATOMIC_RELEASE, __HIP_MEMORY_SCOPE_AGENT);
        } else {
            while (__hip_atomic_load(gen, __ATOMIC_ACQUIRE, __HIP_MEMORY_SCOPE_AGENT) == g)
                __builtin_amdgcn_s_sleep(1);
        }
    }
    __syncthreads();
}

// ---------------------------------------------------------------- persistent kernel
__global__ void __launch_bounds__(TPB)
seqmlp_persistent_kernel(const float* __restrict__ A0,
                         const _Float16* __restrict__ Wsw,   // 4 layers, swizzled
                         const float* __restrict__ W0,       // for W0[:,672:704]
                         const float* __restrict__ b0,
                         const float* __restrict__ b1,
                         const float* __restrict__ b2,
                         const float* __restrict__ b3,
                         const float* __restrict__ b4,
                         const float* __restrict__ W5,
                         const float* __restrict__ b5,
                         float* __restrict__ yout,
                         _Float16* __restrict__ hb0,
                         _Float16* __restrict__ hb1,
                         float* __restrict__ pdot,           // NWG per-WG dot partials
                         unsigned* cnt, unsigned* gen) {
    __shared__ float part[2][4][16];   // [row-tile][k-split][row-in-tile]

    const int tid  = threadIdx.x;
    const int lane = tid & 31;
    const int wave = tid >> 5;
    const int tile = wave & 1;     // 2 row-tiles (32 rows) per WG
    const int ksp  = wave >> 1;    // 4 K-splits of 512
    const int row_tile = blockIdx.x * 2 + tile;   // 0..127
    // A-operand K pattern: lanes 0-15 take K={0..7,16..23}, lanes 16-31 K={8..15,24..31}
    const int aoff = (lane & 16) ? 8 : 0;
    const float* bias[4] = {b1, b2, b3, b4};

    for (int t = 0; t < SEQ_L; ++t) {
        // ---- phase 0: h1 = relu(A0[t] + W0[:,672:] @ ywin + b0) ----
        // y[t-1] is reconstructed here from last step's per-WG dot partials.
        if (tid < WS) {
            int r = blockIdx.x * WS + tid;   // 64 WGs * 32 rows = 2048
            float z = A0[(size_t)t * WIDTH + r] + b0[r];
            const float* wy = W0 + (size_t)r * IN_SZ + WS * NUM_AA;
            float ylast = 0.f;
            if (t > 0) {
                float acc = 0.f;
                #pragma unroll 8
                for (int w = 0; w < NWG; ++w) acc += pdot[w];
                ylast = fmaxf(acc + b5[0], 0.f);
                if (blockIdx.x == 0 && tid == 0) yout[t - 1] = ylast;
            }
            #pragma unroll
            for (int j = 0; j < WS - 1; ++j) {       // y_window[j] = y[t-32+j]
                int si = t - WS + j;
                float yv = (si >= 0) ? yout[si] : 0.f;
                z = fmaf(wy[j], yv, z);
            }
            z = fmaf(wy[WS - 1], ylast, z);          // newest window entry
            hb0[r] = (_Float16)fmaxf(z, 0.f);
        }
        gbarrier(cnt, gen);

        // ---- phases 1..4: WMMA GEMV chain, f16 weights/activations, f32 accum ----
        // Activations ride in the A operand (replicated over M rows), weights in B.
        // Both operands are written ONLY by VMEM loads -> no WMMA operand hazards.
        const _Float16* hin  = hb0;
        _Float16*       hout = hb1;
        for (int l = 0; l < 4; ++l) {
            const _Float16* Wl = Wsw + ((size_t)l << 22);
            v8f c = {};
            const int kc0 = ksp * 16;            // 16 chunks of K=32 -> 512 K per wave
            #pragma unroll
            for (int kc = 0; kc < 16; ++kc) {
                const int kchunk = kc0 + kc;
                // A: h[k0..k0+31] replicated across the 16 M rows
                const _Float16* hp = hin + kchunk * 32 + aoff;
                v8h a0 = *(const v8h*)(hp);
                v8h a1 = *(const v8h*)(hp + 16);
                v16h a = __builtin_shufflevector(a0, a1,
                            0,1,2,3,4,5,6,7,8,9,10,11,12,13,14,15);
                // B: 32x16 weight tile, coalesced 32B/lane from swizzled layout
                v16h b = *(const v16h*)(Wl + ((size_t)(row_tile * 64 + kchunk)) * 512
                                           + (size_t)lane * 16);
                c = __builtin_amdgcn_wmma_f32_16x16x32_f16(
                        false, a, false, b, (short)0, c, false, false);
            }
            // D row M=0 lives in c[0]: lanes 0..15 hold columns N=0..15
            if (lane < 16) part[tile][ksp][lane] = c[0];
            __syncthreads();
            if (tid < 2 * 16) {                   // exactly wave 0
                int lt = tid >> 4, m = tid & 15;
                int R = blockIdx.x * 32 + tid;
                float s = part[lt][0][m] + part[lt][1][m] + part[lt][2][m] + part[lt][3][m]
                        + bias[l][R];
                s = fmaxf(s, 0.f);
                if (l < 3) {
                    hout[R] = (_Float16)s;
                } else {
                    // fused output layer: per-WG partial of W5 . h5 (wave-0 reduce)
                    float pd = W5[R] * s;
                    #pragma unroll
                    for (int o = 16; o > 0; o >>= 1) pd += __shfl_down(pd, o, 32);
                    if (tid == 0) pdot[blockIdx.x] = pd;
                }
            }
            gbarrier(cnt, gen);
            const _Float16* tmp = hin; hin = hout; hout = (_Float16*)tmp;
        }
    }
    // final output: y[SEQ_L-1]
    if (blockIdx.x == 0 && threadIdx.x == 0) {
        float acc = 0.f;
        for (int w = 0; w < NWG; ++w) acc += pdot[w];
        yout[SEQ_L - 1] = fmaxf(acc + b5[0], 0.f);
    }
}

// ---------------------------------------------------------------- launch
extern "C" void kernel_launch(void* const* d_in, const int* in_sizes, int n_in,
                              void* d_out, int out_size, void* d_ws, size_t ws_size,
                              hipStream_t stream) {
    const int*   x  = (const int*)  d_in[0];
    const float* W0 = (const float*)d_in[1];  const float* b0 = (const float*)d_in[2];
    const float* W1 = (const float*)d_in[3];  const float* b1 = (const float*)d_in[4];
    const float* W2 = (const float*)d_in[5];  const float* b2 = (const float*)d_in[6];
    const float* W3 = (const float*)d_in[7];  const float* b3 = (const float*)d_in[8];
    const float* W4 = (const float*)d_in[9];  const float* b4 = (const float*)d_in[10];
    const float* W5 = (const float*)d_in[11]; const float* b5 = (const float*)d_in[12];
    float* yout = (float*)d_out;

    char* ws = (char*)d_ws;
    unsigned* cnt = (unsigned*)ws;
    unsigned* gen = cnt + 1;
    float*    A0  = (float*)(ws + 256);
    const size_t A0_BYTES  = (size_t)SEQ_L * WIDTH * sizeof(float);        // 16 MB
    _Float16* Wsw = (_Float16*)(ws + 256 + A0_BYTES);
    const size_t WSW_BYTES = (size_t)4 * LAYER_ELEMS * sizeof(_Float16);   // 32 MB
    _Float16* hb0 = (_Float16*)(ws + 256 + A0_BYTES + WSW_BYTES);
    _Float16* hb1 = hb0 + WIDTH;
    float*    pdot = (float*)(hb1 + WIDTH);

    init_bar_kernel<<<1, 1, 0, stream>>>(cnt, gen);
    precompute_A0_kernel<<<SEQ_L, TPB, 0, stream>>>(x, W0, A0);
    convert_swizzle_kernel<<<(4 * LAYER_ELEMS) / TPB, TPB, 0, stream>>>(W1, W2, W3, W4, Wsw);
    seqmlp_persistent_kernel<<<NWG, TPB, 0, stream>>>(
        A0, Wsw, W0, b0, b1, b2, b3, b4, W5, b5, yout, hb0, hb1, pdot, cnt, gen);
}
